// ChunkedCrossAttention_65687229825594
// MI455X (gfx1250) — compile-verified, matrix operand order checked by reading, not run on hardware
//
#include <hip/hip_runtime.h>

// ---------------------------------------------------------------------------
// Types
// ---------------------------------------------------------------------------
typedef __attribute__((ext_vector_type(16))) __bf16 bf16x16;
typedef __attribute__((ext_vector_type(8)))  __bf16 bf16x8;
typedef __attribute__((ext_vector_type(8)))  float  f32x8;

#define WMMA_BF16(a, b, c) \
    __builtin_amdgcn_wmma_f32_16x16x32_bf16(false, (a), false, (b), (short)0, (c), false, false)

// Problem constants (match the reference)
#define BATCH 2
#define TQ    2048
#define TC    4096
#define DMODEL 1024
#define NHEADS 16
#define DHEAD  64

// ---------------------------------------------------------------------------
// DPP16 butterfly reductions over the 16-lane groups (wave32: two independent
// groups, exactly matching the WMMA C-layout row striping).
// Masks {1,2,7,15}: after xor1+xor2 each quad is uniform, so ROW_HALF_MIRROR
// (lane^7) acts as the stride-4 stage and ROW_MIRROR (lane^15) as stride-8.
// All VALU — no ds_bpermute, no DScnt waits.
// ---------------------------------------------------------------------------
#define DPP_QUAD_XOR1 0xB1   // quad_perm [1,0,3,2]
#define DPP_QUAD_XOR2 0x4E   // quad_perm [2,3,0,1]
#define DPP_ROW_HMIR  0x141  // row_half_mirror (lane ^ 7)
#define DPP_ROW_MIR   0x140  // row_mirror      (lane ^ 15)

static __device__ __forceinline__ float dpp_bcast_f32(float x, int ctrl_unused) { return x; }

static __device__ __forceinline__ float dpp_step(float x, const int ctrl_id)
{
    int xi = __builtin_bit_cast(int, x);
    int yi;
    switch (ctrl_id) {
    case 0: yi = __builtin_amdgcn_update_dpp(xi, xi, DPP_QUAD_XOR1, 0xF, 0xF, true); break;
    case 1: yi = __builtin_amdgcn_update_dpp(xi, xi, DPP_QUAD_XOR2, 0xF, 0xF, true); break;
    case 2: yi = __builtin_amdgcn_update_dpp(xi, xi, DPP_ROW_HMIR,  0xF, 0xF, true); break;
    default: yi = __builtin_amdgcn_update_dpp(xi, xi, DPP_ROW_MIR,  0xF, 0xF, true); break;
    }
    return __builtin_bit_cast(float, yi);
}

static __device__ __forceinline__ float row_max16(float x)
{
    x = fmaxf(x, dpp_step(x, 0));
    x = fmaxf(x, dpp_step(x, 1));
    x = fmaxf(x, dpp_step(x, 2));
    x = fmaxf(x, dpp_step(x, 3));
    return x;
}

static __device__ __forceinline__ float row_sum16(float x)
{
    x += dpp_step(x, 0);
    x += dpp_step(x, 1);
    x += dpp_step(x, 2);
    x += dpp_step(x, 3);
    return x;
}

// ---------------------------------------------------------------------------
// Fragment loader.
// 16-bit A (16x32, MxK) per CDNA5 ISA: lane m (0..15) holds K {0..7, 16..23},
// lane m+16 holds K {8..15, 24..31}; i.e. two contiguous 16B chunks per lane.
// The B fragment (32x16, KxN) mirrors this with N in place of M, so an NT
// GEMM (both operands K-contiguous) loads both fragments with this routine.
// ---------------------------------------------------------------------------
static __device__ __forceinline__ bf16x16
load_frag(const __bf16* base, int ld, int row, int k0, int lane)
{
    const int m = lane & 15;
    const int h = lane >> 4;
    const __bf16* p = base + (size_t)(row + m) * (size_t)ld + k0 + h * 8;
    union { bf16x16 v; bf16x8 q[2]; } u;
    u.q[0] = *(const bf16x8*)(p);         // K = k0 + h*8 .. +7
    u.q[1] = *(const bf16x8*)(p + 16);    // K = k0 + 16 + h*8 .. +7
    return u.v;
}

// ---------------------------------------------------------------------------
// f32 -> bf16 cast
// ---------------------------------------------------------------------------
__global__ void cast_f32_to_bf16(const float* __restrict__ x,
                                 __bf16* __restrict__ y, int n)
{
    int i = blockIdx.x * blockDim.x + threadIdx.x;
    if (i < n) y[i] = (__bf16)x[i];
}

// ---------------------------------------------------------------------------
// NT GEMM:  C[M, 1024] = A[M, K](bf16) @ W[1024, K](bf16)^T, f32 accumulate.
// MODE 0: store f32 row-major [M,1024]                (final output proj)
// MODE 1: store bf16 head-major   [B, H, T, 64]       (Q, K)
// MODE 2: store bf16 head-transp. [B, H, 64, T]       (V)
// Block: 256 threads (8 waves), tile 128x128; wave tile 32x64.
// ---------------------------------------------------------------------------
template<int MODE>
__global__ __launch_bounds__(256) void gemm_nt_bf16(
    const __bf16* __restrict__ A, const __bf16* __restrict__ W,
    float* __restrict__ Cf, __bf16* __restrict__ Cb,
    int M, int K, int T)
{
    const int NT = DMODEL / 128;
    const int tm = (blockIdx.x / NT) * 128;
    const int tn = (blockIdx.x % NT) * 128;
    const int lane = threadIdx.x & 31;
    const int w    = threadIdx.x >> 5;
    const int wm   = (w & 3) * 32;
    const int wn   = (w >> 2) * 64;

    f32x8 acc[2][4];
#pragma unroll
    for (int i = 0; i < 2; ++i)
#pragma unroll
        for (int j = 0; j < 4; ++j)
            acc[i][j] = {};

    for (int kk = 0; kk < K; kk += 32) {
        bf16x16 af[2], bf[4];
#pragma unroll
        for (int i = 0; i < 2; ++i)
            af[i] = load_frag(A, K, tm + wm + i * 16, kk, lane);
#pragma unroll
        for (int j = 0; j < 4; ++j)
            bf[j] = load_frag(W, K, tn + wn + j * 16, kk, lane);
#pragma unroll
        for (int i = 0; i < 2; ++i)
#pragma unroll
            for (int j = 0; j < 4; ++j)
                acc[i][j] = WMMA_BF16(af[i], bf[j], acc[i][j]);
    }

    const int nl   = lane & 15;
    const int half = lane >> 4;
#pragma unroll
    for (int i = 0; i < 2; ++i)
#pragma unroll
        for (int j = 0; j < 4; ++j)
#pragma unroll
            for (int g = 0; g < 8; ++g) {
                const int row = tm + wm + i * 16 + g + half * 8;
                const int col = tn + wn + j * 16 + nl;
                const float v = acc[i][j][g];
                if (MODE == 0) {
                    Cf[(size_t)row * DMODEL + col] = v;
                } else {
                    const int bb = row / T;
                    const int t  = row - bb * T;
                    const int hh = col >> 6;
                    const int d  = col & 63;
                    size_t idx;
                    if (MODE == 1)
                        idx = ((size_t)(bb * NHEADS + hh) * T + t) * DHEAD + d;
                    else
                        idx = ((size_t)(bb * NHEADS + hh) * DHEAD + d) * (size_t)T + t;
                    Cb[idx] = (__bf16)v;
                }
            }
}

// ---------------------------------------------------------------------------
// Flash attention (non-causal), one block per (b, h, 64-row q tile).
// 4 waves x 16 q-rows. Keys processed in chunks of 64 with online softmax
// (16 WMMAs per chunk: 8 for S = Q K^T, 8 for O += P V).
// Q:  [B,H,TQ,64] bf16   K: [B,H,TC,64] bf16   Vt: [B,H,64,TC] bf16
// O:  [B,TQ,1024] bf16 (row-major, feeds the final projection GEMM)
// ---------------------------------------------------------------------------
__global__ __launch_bounds__(128) void flash_attn_bf16(
    const __bf16* __restrict__ Q, const __bf16* __restrict__ Km,
    const __bf16* __restrict__ Vt, __bf16* __restrict__ O)
{
    __shared__ __bf16 ldsP[4][16][64];           // per-wave P staging (2KB/wave)

    const int tile = blockIdx.x;
    const int qb = tile & 31;                    // TQ/64 = 32 q tiles
    const int hh = (tile >> 5) & 15;
    const int bb = tile >> 9;
    const int lane = threadIdx.x & 31;
    const int w    = threadIdx.x >> 5;
    const int q0   = qb * 64 + w * 16;

    const __bf16* qbase = Q  + ((size_t)(bb * NHEADS + hh) * TQ + q0) * DHEAD;
    const __bf16* kbase = Km + (size_t)(bb * NHEADS + hh) * TC * DHEAD;
    const __bf16* vbase = Vt + (size_t)(bb * NHEADS + hh) * DHEAD * TC;

    bf16x16 qa[2];
    qa[0] = load_frag(qbase, DHEAD, 0, 0,  lane);
    qa[1] = load_frag(qbase, DHEAD, 0, 32, lane);

    f32x8 o[4];
#pragma unroll
    for (int t = 0; t < 4; ++t) o[t] = {};
    float mrow[8], lrow[8];
#pragma unroll
    for (int g = 0; g < 8; ++g) { mrow[g] = -1e30f; lrow[g] = 0.0f; }

    const float scale = 0.125f;                  // 1/sqrt(64)
    const int nl = lane & 15;

    for (int kc = 0; kc < TC; kc += 64) {
        // --- S = Q @ K^T : 16 x 64 score tile (four 16x16 C tiles) ---
        f32x8 s[4];
#pragma unroll
        for (int j = 0; j < 4; ++j) s[j] = {};
#pragma unroll
        for (int dc = 0; dc < 2; ++dc)
#pragma unroll
            for (int j = 0; j < 4; ++j) {
                bf16x16 kf = load_frag(kbase, DHEAD, kc + j * 16, dc * 32, lane);
                s[j] = WMMA_BF16(qa[dc], kf, s[j]);
            }

        // --- online softmax per row (DPP16 butterflies, pure VALU) ---
#pragma unroll
        for (int g = 0; g < 8; ++g) {
            float a0 = s[0][g] * scale;
            float a1 = s[1][g] * scale;
            float a2 = s[2][g] * scale;
            float a3 = s[3][g] * scale;
            float mx = row_max16(fmaxf(fmaxf(a0, a1), fmaxf(a2, a3)));
            const float mn   = fmaxf(mrow[g], mx);
            const float corr = __expf(mrow[g] - mn);
            const float p0 = __expf(a0 - mn);
            const float p1 = __expf(a1 - mn);
            const float p2 = __expf(a2 - mn);
            const float p3 = __expf(a3 - mn);
            const float rs = row_sum16((p0 + p1) + (p2 + p3));
            lrow[g] = lrow[g] * corr + rs;
            mrow[g] = mn;
#pragma unroll
            for (int t = 0; t < 4; ++t) o[t][g] *= corr;

            const int Mr = g + ((lane >> 4) << 3);      // C-layout row
            ldsP[w][Mr][nl]      = (__bf16)p0;          // keys kc .. kc+15
            ldsP[w][Mr][nl + 16] = (__bf16)p1;
            ldsP[w][Mr][nl + 32] = (__bf16)p2;
            ldsP[w][Mr][nl + 48] = (__bf16)p3;
        }
        asm volatile("" ::: "memory");                  // keep DS store->load order

        // --- O += P @ V  (P reloaded from LDS in A-fragment layout) ---
        bf16x16 pf0 = load_frag(&ldsP[w][0][0], 64, 0, 0,  lane);
        bf16x16 pf1 = load_frag(&ldsP[w][0][0], 64, 0, 32, lane);
#pragma unroll
        for (int t = 0; t < 4; ++t) {
            bf16x16 vf0 = load_frag(vbase, TC, t * 16, kc,      lane);
            bf16x16 vf1 = load_frag(vbase, TC, t * 16, kc + 32, lane);
            o[t] = WMMA_BF16(pf0, vf0, o[t]);
            o[t] = WMMA_BF16(pf1, vf1, o[t]);
        }
    }

    // --- normalize and write O in [B, TQ, D] row-major bf16 ---
    const int half = lane >> 4;
#pragma unroll
    for (int t = 0; t < 4; ++t)
#pragma unroll
        for (int g = 0; g < 8; ++g) {
            const int Mr = g + half * 8;
            const size_t row = (size_t)bb * TQ + q0 + Mr;
            const int col = hh * DHEAD + t * 16 + nl;
            O[row * DMODEL + col] = (__bf16)(o[t][g] / lrow[g]);
        }
}

// ---------------------------------------------------------------------------
// Host-side orchestration
// ---------------------------------------------------------------------------
extern "C" void kernel_launch(void* const* d_in, const int* in_sizes, int n_in,
                              void* d_out, int out_size, void* d_ws, size_t ws_size,
                              hipStream_t stream)
{
    const float* query   = (const float*)d_in[0];  // [2, 2048, 1024]
    const float* context = (const float*)d_in[1];  // [2, 4096, 1024]
    const float* Wq      = (const float*)d_in[2];  // [1024, 1024]
    const float* Wk      = (const float*)d_in[3];
    const float* Wv      = (const float*)d_in[4];
    const float* Wo      = (const float*)d_in[5];
    float* out = (float*)d_out;                    // [2, 2048, 1024] f32

    const size_t nQ  = (size_t)BATCH * TQ * DMODEL;
    const size_t nC  = (size_t)BATCH * TC * DMODEL;
    const size_t nW  = (size_t)DMODEL * DMODEL;

    char* ws = (char*)d_ws;
    __bf16* qh  = (__bf16*)(ws);                                   size_t off = nQ * 2;
    __bf16* ch  = (__bf16*)(ws + off);                             off += nC * 2;
    __bf16* wqh = (__bf16*)(ws + off);                             off += nW * 2;
    __bf16* wkh = (__bf16*)(ws + off);                             off += nW * 2;
    __bf16* wvh = (__bf16*)(ws + off);                             off += nW * 2;
    __bf16* woh = (__bf16*)(ws + off);                             off += nW * 2;
    __bf16* Qb  = (__bf16*)(ws + off);                             off += nQ * 2;  // [B,H,TQ,64]
    __bf16* Kb  = (__bf16*)(ws + off);                             off += nC * 2;  // [B,H,TC,64]
    __bf16* Vtb = (__bf16*)(ws + off);                             off += nC * 2;  // [B,H,64,TC]
    __bf16* Ob  = (__bf16*)(ws + off);                             off += nQ * 2;  // [B,TQ,D]

    // 1) casts to bf16
    cast_f32_to_bf16<<<(int)((nQ + 255) / 256), 256, 0, stream>>>(query,   qh,  (int)nQ);
    cast_f32_to_bf16<<<(int)((nC + 255) / 256), 256, 0, stream>>>(context, ch,  (int)nC);
    cast_f32_to_bf16<<<(int)((nW + 255) / 256), 256, 0, stream>>>(Wq,      wqh, (int)nW);
    cast_f32_to_bf16<<<(int)((nW + 255) / 256), 256, 0, stream>>>(Wk,      wkh, (int)nW);
    cast_f32_to_bf16<<<(int)((nW + 255) / 256), 256, 0, stream>>>(Wv,      wvh, (int)nW);
    cast_f32_to_bf16<<<(int)((nW + 255) / 256), 256, 0, stream>>>(Wo,      woh, (int)nW);

    // 2) projections (WMMA GEMMs)
    const int MQ = BATCH * TQ;    // 4096
    const int MC = BATCH * TC;    // 8192
    const int NTILES = DMODEL / 128;
    gemm_nt_bf16<1><<<(MQ / 128) * NTILES, 256, 0, stream>>>(qh, wqh, nullptr, Qb,  MQ, DMODEL, TQ);
    gemm_nt_bf16<1><<<(MC / 128) * NTILES, 256, 0, stream>>>(ch, wkh, nullptr, Kb,  MC, DMODEL, TC);
    gemm_nt_bf16<2><<<(MC / 128) * NTILES, 256, 0, stream>>>(ch, wvh, nullptr, Vtb, MC, DMODEL, TC);

    // 3) attention: B*H*(TQ/64) blocks, 4 waves each
    flash_attn_bf16<<<BATCH * NHEADS * (TQ / 64), 128, 0, stream>>>(Qb, Kb, Vtb, Ob);

    // 4) output projection -> f32 d_out
    gemm_nt_bf16<0><<<(MQ / 128) * NTILES, 256, 0, stream>>>(Ob, woh, out, nullptr, MQ, DMODEL, TQ);
}
